// cross_attention_214748365142
// MI455X (gfx1250) — matrix operand order repaired and checked by reading, not compiled, not run on hardware
//
#include <hip/hip_runtime.h>

typedef __attribute__((ext_vector_type(16))) __bf16 v16bf;
typedef __attribute__((ext_vector_type(8)))  float  v8f;

#define BQ 8
#define CC 64
#define HH 256
#define WW 256
#define PITCH_QK 264   // bf16 elems, multiple of 8 (16B-aligned vector loads)
#define PITCH_T  72    // bf16 elems, multiple of 8
#define QKV_ELEMS 33554432ull   // 8*256*64*256

#define SMEM_Q    147200
#define SMEM_KV   159232
#define SMEM_ATTN 113664

#define V8F_ZERO ((v8f){0.f, 0.f, 0.f, 0.f, 0.f, 0.f, 0.f, 0.f})

__device__ __forceinline__ unsigned short f2bf(float f) {
  union { float f; unsigned u; } v; v.f = f;
  unsigned r = v.u + 0x7FFFu + ((v.u >> 16) & 1u);   // round-to-nearest-even
  return (unsigned short)(r >> 16);
}
__device__ __forceinline__ float bf2f(unsigned short s) {
  union { unsigned u; float f; } v; v.u = ((unsigned)s) << 16; return v.f;
}

// A fragment (16x32 bf16) from row-major [m][k] ushort LDS (pitch multiple of 8).
// ISA 7.12.2: lane L holds row (L&15); VGPR0-3 hold K = kb..kb+7, VGPR4-7 hold
// K = kb+16..kb+23, where kb = 8 for lanes 16-31.
__device__ __forceinline__ v16bf load_a_frag(const unsigned short* base, int pitch,
                                             int m0, int k0, int lane) {
  int row = m0 + (lane & 15);
  int kb  = k0 + ((lane & 16) ? 8 : 0);
  const unsigned short* p = base + row * pitch + kb;
  uint4 lo = *(const uint4*)(p);
  uint4 hi = *(const uint4*)(p + 16);
  v16bf a;
  __builtin_memcpy(&a, &lo, 16);
  __builtin_memcpy(((char*)&a) + 16, &hi, 16);
  return a;
}

// B fragment (32x16 bf16) where storage is R[n][k] (i.e. B[k][n] = R[n][k]).
// Lane L holds column n=(L&15); K = kb..kb+7 and kb+16..kb+23, kb += 8 for lanes 16-31.
__device__ __forceinline__ v16bf load_b_frag(const unsigned short* base, int pitch,
                                             int n0, int k0, int lane) {
  int row = n0 + (lane & 15);
  int kb  = k0 + ((lane & 16) ? 8 : 0);
  const unsigned short* p = base + row * pitch + kb;
  uint4 lo = *(const uint4*)(p);
  uint4 hi = *(const uint4*)(p + 16);
  v16bf b;
  __builtin_memcpy(&b, &lo, 16);
  __builtin_memcpy(((char*)&b) + 16, &hi, 16);
  return b;
}

__device__ __forceinline__ v8f wmma_bf16(v16bf a, v16bf b, v8f c) {
  return __builtin_amdgcn_wmma_f32_16x16x32_bf16(false, a, false, b, (short)0, c,
                                                 false, false);
}

// Stage 3 input rows (h-1,h,h+1) of one image as bf16 into sIN[(c*3+r)*256 + w].
__device__ __forceinline__ void stage_rows(const float* __restrict__ inb, int h,
                                           unsigned short* sIN, int t) {
#pragma unroll
  for (int it = 0; it < 48; ++it) {
    int g = t + 256 * it;                       // 12288 float4 groups
    int c = g / 192; int rem = g - c * 192;
    int r = rem >> 6; int wv = rem & 63;
    int hh = h - 1 + r;
    float4 val = {0.f, 0.f, 0.f, 0.f};
    if ((unsigned)hh < 256u) val = *(const float4*)(inb + c * 65536 + hh * 256 + wv * 4);
    unsigned short* d = sIN + (c * 3 + r) * 256 + wv * 4;
    d[0] = f2bf(val.x); d[1] = f2bf(val.y); d[2] = f2bf(val.z); d[3] = f2bf(val.w);
  }
}

// Depthwise 3x3 (+bias): thread t owns column w=t for all 64 channels -> sT[w][c].
__device__ __forceinline__ void depthwise(const unsigned short* sIN, const float* sDW,
                                          const float* sDB, unsigned short* sT, int t) {
  int w = t;
  for (int c = 0; c < CC; ++c) {
    float acc = sDB[c];
    const float* wt = sDW + c * 9;
#pragma unroll
    for (int r = 0; r < 3; ++r) {
      const unsigned short* rowp = sIN + (c * 3 + r) * 256;
      float x1 = bf2f(rowp[w]);
      float x0 = (w > 0)   ? bf2f(rowp[w - 1]) : 0.f;
      float x2 = (w < 255) ? bf2f(rowp[w + 1]) : 0.f;
      acc += x0 * wt[r * 3 + 0] + x1 * wt[r * 3 + 1] + x2 * wt[r * 3 + 2];
    }
    sT[w * PITCH_T + c] = f2bf(acc);
  }
}

// ---------------------------------------------------------------------------
// Q projection: depthwise+pointwise, one image row per workgroup.
// Output bf16 row-major [b][h][c][w].
// ---------------------------------------------------------------------------
extern "C" __global__ __launch_bounds__(256)
void proj_q_kernel(const float* __restrict__ in, const float* __restrict__ dw,
                   const float* __restrict__ db, const float* __restrict__ pw,
                   const float* __restrict__ pb, unsigned short* __restrict__ outg) {
  extern __shared__ __align__(16) char smem[];
  unsigned short* sIN = (unsigned short*)smem;              // 98304B
  unsigned short* sT  = (unsigned short*)(smem + 98304);    // [w*72 + c] 36864B
  unsigned short* sPW = (unsigned short*)(smem + 135168);   // [o*72 + c]  9216B
  float* sDW = (float*)(smem + 144384);
  float* sDB = (float*)(smem + 146688);
  float* sPB = (float*)(smem + 146944);
  unsigned short* sO = sIN;  // sIN dead after depthwise

  int t = threadIdx.x;
  int h = blockIdx.x;
  int b = blockIdx.y;
  const float* inb = in + (size_t)b * (CC * HH * WW);

  stage_rows(inb, h, sIN, t);
#pragma unroll
  for (int it = 0; it < 16; ++it) {
    int f = t + 256 * it; int o = f >> 6; int c = f & 63;
    sPW[o * PITCH_T + c] = f2bf(pw[f]);
  }
  for (int it = 0; it < 3; ++it) { int f = t + 256 * it; if (f < 576) sDW[f] = dw[f]; }
  if (t < 64) { sDB[t] = db[t]; sPB[t] = pb[t]; }
  __syncthreads();

  depthwise(sIN, sDW, sDB, sT, t);
  __syncthreads();

  int lane = t & 31;
  int wid  = t >> 5;                            // 8 waves
  int m0 = (wid >> 1) * 16;
  int nh = wid & 1;
  int hi8 = (lane & 16) ? 8 : 0;
  v16bf a0 = load_a_frag(sPW, PITCH_T, m0, 0, lane);
  v16bf a1 = load_a_frag(sPW, PITCH_T, m0, 32, lane);
#pragma unroll
  for (int nt = 0; nt < 8; ++nt) {
    int n0 = nh * 128 + nt * 16;
    v16bf b0 = load_b_frag(sT, PITCH_T, n0, 0, lane);
    v16bf b1 = load_b_frag(sT, PITCH_T, n0, 32, lane);
    v8f acc = V8F_ZERO;
    acc = wmma_bf16(a0, b0, acc);
    acc = wmma_bf16(a1, b1, acc);
    int wcol = n0 + (lane & 15);
#pragma unroll
    for (int r = 0; r < 8; ++r) {
      int o = m0 + r + hi8;
      sO[o * PITCH_QK + wcol] = f2bf(acc[r] + sPB[o]);
    }
  }
  __syncthreads();

  size_t outb = ((size_t)b * HH + h) * (CC * WW);
#pragma unroll
  for (int it = 0; it < 8; ++it) {
    int g = t + 256 * it;
    int c = g >> 5; int w0 = (g & 31) * 8;
    *(uint4*)(outg + outb + c * 256 + w0) = *(const uint4*)(sO + c * PITCH_QK + w0);
  }
}

// ---------------------------------------------------------------------------
// Fused K+V projection: stage ctx rows ONCE, run both depthwise+pointwise
// passes. K: row-major [b][h][d][w] (staged in sT after it is dead, K tiles
// held in VGPRs across the barrier). V: transposed [b][h][w][c] (staged in the
// then-dead sIN region).
// ---------------------------------------------------------------------------
extern "C" __global__ __launch_bounds__(256)
void proj_kv_kernel(const float* __restrict__ in,
                    const float* __restrict__ kdw, const float* __restrict__ kdb,
                    const float* __restrict__ kpw, const float* __restrict__ kpb,
                    const float* __restrict__ vdw, const float* __restrict__ vdb,
                    const float* __restrict__ vpw, const float* __restrict__ vpb,
                    unsigned short* __restrict__ Kg, unsigned short* __restrict__ Vtg) {
  extern __shared__ __align__(16) char smem[];
  unsigned short* sIN  = (unsigned short*)smem;              // 98304B
  unsigned short* sT   = (unsigned short*)(smem + 98304);    // 36864B (>= 33792 for sOK)
  unsigned short* sPWk = (unsigned short*)(smem + 135168);   // 9216B
  unsigned short* sPWv = (unsigned short*)(smem + 144384);   // 9216B
  float* sDWk = (float*)(smem + 153600);
  float* sDWv = (float*)(smem + 155904);
  float* sDBk = (float*)(smem + 158208);
  float* sPBk = (float*)(smem + 158464);
  float* sDBv = (float*)(smem + 158720);
  float* sPBv = (float*)(smem + 158976);

  int t = threadIdx.x;
  int h = blockIdx.x;
  int b = blockIdx.y;
  const float* inb = in + (size_t)b * (CC * HH * WW);
  size_t outb = ((size_t)b * HH + h) * (CC * WW);

  stage_rows(inb, h, sIN, t);
#pragma unroll
  for (int it = 0; it < 16; ++it) {
    int f = t + 256 * it; int o = f >> 6; int c = f & 63;
    sPWk[o * PITCH_T + c] = f2bf(kpw[f]);
    sPWv[o * PITCH_T + c] = f2bf(vpw[f]);
  }
  for (int it = 0; it < 3; ++it) {
    int f = t + 256 * it;
    if (f < 576) { sDWk[f] = kdw[f]; sDWv[f] = vdw[f]; }
  }
  if (t < 64) { sDBk[t] = kdb[t]; sPBk[t] = kpb[t]; sDBv[t] = vdb[t]; sPBv[t] = vpb[t]; }
  __syncthreads();

  int lane = t & 31;
  int wid  = t >> 5;
  int m0 = (wid >> 1) * 16;
  int nh = wid & 1;
  int hi8 = (lane & 16) ? 8 : 0;

  // ---- K pass ----
  depthwise(sIN, sDWk, sDBk, sT, t);
  __syncthreads();

  {
    v16bf a0 = load_a_frag(sPWk, PITCH_T, m0, 0, lane);
    v16bf a1 = load_a_frag(sPWk, PITCH_T, m0, 32, lane);
    v8f accK[8];
#pragma unroll
    for (int nt = 0; nt < 8; ++nt) {
      int n0 = nh * 128 + nt * 16;
      v16bf b0 = load_b_frag(sT, PITCH_T, n0, 0, lane);
      v16bf b1 = load_b_frag(sT, PITCH_T, n0, 32, lane);
      v8f acc = V8F_ZERO;
      acc = wmma_bf16(a0, b0, acc);
      acc = wmma_bf16(a1, b1, acc);
      accK[nt] = acc;
    }
    __syncthreads();                 // all waves done reading sT -> reuse as sOK
    unsigned short* sOK = sT;        // [c][w], 64*264 elems = 33792B
#pragma unroll
    for (int nt = 0; nt < 8; ++nt) {
      int wcol = nh * 128 + nt * 16 + (lane & 15);
#pragma unroll
      for (int r = 0; r < 8; ++r) {
        int o = m0 + r + hi8;
        sOK[o * PITCH_QK + wcol] = f2bf(accK[nt][r] + sPBk[o]);
      }
    }
    __syncthreads();
#pragma unroll
    for (int it = 0; it < 8; ++it) {
      int g = t + 256 * it;
      int c = g >> 5; int w0 = (g & 31) * 8;
      *(uint4*)(Kg + outb + c * 256 + w0) = *(const uint4*)(sOK + c * PITCH_QK + w0);
    }
    __syncthreads();                 // copy done before V depthwise overwrites sT
  }

  // ---- V pass ----
  depthwise(sIN, sDWv, sDBv, sT, t);
  __syncthreads();                   // sIN now dead -> reuse as sOV

  {
    unsigned short* sOV = sIN;       // [w][c], 256*72 elems = 36864B
    v16bf a0 = load_a_frag(sPWv, PITCH_T, m0, 0, lane);
    v16bf a1 = load_a_frag(sPWv, PITCH_T, m0, 32, lane);
#pragma unroll
    for (int nt = 0; nt < 8; ++nt) {
      int n0 = nh * 128 + nt * 16;
      v16bf b0 = load_b_frag(sT, PITCH_T, n0, 0, lane);
      v16bf b1 = load_b_frag(sT, PITCH_T, n0, 32, lane);
      v8f acc = V8F_ZERO;
      acc = wmma_bf16(a0, b0, acc);
      acc = wmma_bf16(a1, b1, acc);
      int wcol = n0 + (lane & 15);
#pragma unroll
      for (int r = 0; r < 8; ++r) {
        int o = m0 + r + hi8;
        sOV[wcol * PITCH_T + o] = f2bf(acc[r] + sPBv[o]);
      }
    }
    __syncthreads();
#pragma unroll
    for (int it = 0; it < 8; ++it) {
      int g = t + 256 * it;
      int w = g >> 3; int c0 = (g & 7) * 8;
      *(uint4*)(Vtg + outb + w * 64 + c0) = *(const uint4*)(sOV + w * PITCH_T + c0);
    }
  }
}

// ---------------------------------------------------------------------------
// Attention: one (b,h) per workgroup (4 waves). S = Q K^T (64x64) in registers,
// wave-local softmax via shfl_xor over 16-lane row groups, P -> LDS (A layout),
// O = P * V^T via WMMA, fp32 out[b][c][h][w].
// ---------------------------------------------------------------------------
extern "C" __global__ __launch_bounds__(128)
void attn_kernel(const unsigned short* __restrict__ Qg,
                 const unsigned short* __restrict__ Kg,
                 const unsigned short* __restrict__ Vtg,
                 float* __restrict__ outg) {
  extern __shared__ __align__(16) char smem[];
  unsigned short* sQ  = (unsigned short*)smem;              // [c*264 + w] 33792B
  unsigned short* sK  = (unsigned short*)(smem + 33792);    // [d*264 + w] 33792B
  unsigned short* sVt = (unsigned short*)(smem + 67584);    // [w*72 + c]  36864B
  unsigned short* sP  = (unsigned short*)(smem + 104448);   // [c*72 + d]   9216B

  int t = threadIdx.x;
  int h = blockIdx.x;
  int b = blockIdx.y;
  size_t base = ((size_t)b * HH + h) * (CC * WW);

#pragma unroll
  for (int it = 0; it < 16; ++it) {             // 2048 vec8 groups each
    int g = t + 128 * it;
    int c = g >> 5; int w0 = (g & 31) * 8;
    *(uint4*)(sQ + c * PITCH_QK + w0) = *(const uint4*)(Qg + base + g * 8);
    *(uint4*)(sK + c * PITCH_QK + w0) = *(const uint4*)(Kg + base + g * 8);
    int w = g >> 3; int c0 = (g & 7) * 8;
    *(uint4*)(sVt + w * PITCH_T + c0) = *(const uint4*)(Vtg + base + g * 8);
  }
  __syncthreads();

  int lane = t & 31;
  int wid  = t >> 5;                            // 4 waves, 16-row strips
  int m0 = wid * 16;
  int hi8 = (lane & 16) ? 8 : 0;

  // S = Q K^T  (K-dim = w = 256 -> 8 k-steps of 32)
  v8f s[4];
#pragma unroll
  for (int j = 0; j < 4; ++j) s[j] = V8F_ZERO;
  for (int kk = 0; kk < 8; ++kk) {
    int k0 = kk * 32;
    v16bf a = load_a_frag(sQ, PITCH_QK, m0, k0, lane);
#pragma unroll
    for (int j = 0; j < 4; ++j) {
      v16bf bf = load_b_frag(sK, PITCH_QK, j * 16, k0, lane);
      s[j] = wmma_bf16(a, bf, s[j]);
    }
  }

  // Softmax over d=64 per row; scale = 1/sqrt(HEAD_SIZE=8).
  const float scale = 0.35355339059327373f;
#pragma unroll
  for (int r = 0; r < 8; ++r) {
    float m = fmaxf(fmaxf(s[0][r], s[1][r]), fmaxf(s[2][r], s[3][r]));
#pragma unroll
    for (int off = 1; off < 16; off <<= 1) m = fmaxf(m, __shfl_xor(m, off, 32));
    float e[4]; float sum = 0.f;
#pragma unroll
    for (int j = 0; j < 4; ++j) { e[j] = __expf((s[j][r] - m) * scale); sum += e[j]; }
#pragma unroll
    for (int off = 1; off < 16; off <<= 1) sum += __shfl_xor(sum, off, 32);
    float inv = 1.0f / sum;
    int crow = m0 + r + hi8;
#pragma unroll
    for (int j = 0; j < 4; ++j)
      sP[crow * PITCH_T + j * 16 + (lane & 15)] = f2bf(e[j] * inv);
  }
  __syncthreads();

  // O = P * V^T  (K-dim = d = 64 -> 2 k-steps of 32), direct fp32 store.
  v16bf pa0 = load_a_frag(sP, PITCH_T, m0, 0, lane);
  v16bf pa1 = load_a_frag(sP, PITCH_T, m0, 32, lane);
  for (int n = 0; n < 16; ++n) {
    int n0 = n * 16;
    v16bf b0 = load_b_frag(sVt, PITCH_T, n0, 0, lane);
    v16bf b1 = load_b_frag(sVt, PITCH_T, n0, 32, lane);
    v8f o = V8F_ZERO;
    o = wmma_bf16(pa0, b0, o);
    o = wmma_bf16(pa1, b1, o);
    int w = n0 + (lane & 15);
#pragma unroll
    for (int r = 0; r < 8; ++r) {
      int c = m0 + r + hi8;
      outg[(((size_t)b * CC + c) * HH + h) * WW + w] = o[r];
    }
  }
}

extern "C" void kernel_launch(void* const* d_in, const int* in_sizes, int n_in,
                              void* d_out, int out_size, void* d_ws, size_t ws_size,
                              hipStream_t stream) {
  (void)in_sizes; (void)n_in; (void)out_size; (void)ws_size;
  const float* hidden = (const float*)d_in[0];
  const float* ctx    = (const float*)d_in[1];
  const float* q_dw = (const float*)d_in[2];
  const float* q_db = (const float*)d_in[3];
  const float* q_pw = (const float*)d_in[4];
  const float* q_pb = (const float*)d_in[5];
  const float* k_dw = (const float*)d_in[6];
  const float* k_db = (const float*)d_in[7];
  const float* k_pw = (const float*)d_in[8];
  const float* k_pb = (const float*)d_in[9];
  const float* v_dw = (const float*)d_in[10];
  const float* v_db = (const float*)d_in[11];
  const float* v_pw = (const float*)d_in[12];
  const float* v_pb = (const float*)d_in[13];

  unsigned short* Qg  = (unsigned short*)d_ws;          // bf16 [b][h][c][w]
  unsigned short* Kg  = Qg + QKV_ELEMS;                 // bf16 [b][h][d][w]
  unsigned short* Vtg = Kg + QKV_ELEMS;                 // bf16 [b][h][w][c]

  (void)hipFuncSetAttribute((const void*)proj_q_kernel,
                            hipFuncAttributeMaxDynamicSharedMemorySize, SMEM_Q);
  (void)hipFuncSetAttribute((const void*)proj_kv_kernel,
                            hipFuncAttributeMaxDynamicSharedMemorySize, SMEM_KV);
  (void)hipFuncSetAttribute((const void*)attn_kernel,
                            hipFuncAttributeMaxDynamicSharedMemorySize, SMEM_ATTN);

  dim3 grid(HH, BQ);
  proj_q_kernel<<<grid, 256, SMEM_Q, stream>>>(hidden, q_dw, q_db, q_pw, q_pb, Qg);
  proj_kv_kernel<<<grid, 256, SMEM_KV, stream>>>(ctx, k_dw, k_db, k_pw, k_pb,
                                                 v_dw, v_db, v_pw, v_pb, Kg, Vtg);
  attn_kernel<<<grid, 128, SMEM_ATTN, stream>>>(Qg, Kg, Vtg, (float*)d_out);
}
